// SimpleGNNLayer_16329465659892
// MI455X (gfx1250) — matrix operand order repaired and checked by reading, not compiled, not run on hardware
//
#include <hip/hip_runtime.h>
#include <hip/hip_fp16.h>
#include <cstdint>

// ---------------------------------------------------------------------------
// SimpleGNNLayer for MI455X (gfx1250, wave32, WMMA)
//   T=50000 nodes, D=128, E=1.6M edges, DE=3
//   edge MLP = GELU(concat(H[src], ea) @ W1 + b1) @ W2 + b2, scatter-add to dst,
//   then LayerNorm(H + agg).
// ---------------------------------------------------------------------------

#define T_NODES 50000
#define D_FEAT  128
#define E_EDGES 1600000
#define DE_ATTR 3
#define K1_PAD  160      // 131 padded to 5 * 32
#define TILE_M  32       // edges per block tile (2 WMMA M-tiles)
#define LN_EPS  1e-5f

typedef __attribute__((ext_vector_type(16))) _Float16 v16h;
typedef __attribute__((ext_vector_type(8)))  float    v8f;

union FragH {            // 16 halves = two 128-bit LDS loads
    v16h  h;
    uint4 u[2];
};

// GELU, tanh form; tanh error (~1e-4) is below the f16 quantization of the
// GEMM2 input, and gfx1250 has a hardware v_tanh_f32 transcendental.
__device__ __forceinline__ float gelu_fast(float x) {
    // 0.79788456*(x + 0.044715 x^3) = x*(0.79788456 + 0.03567740814 x^2)
    float u = x * (0.7978845608028654f + 0.03567740814183f * x * x);
    float t;
#if __has_builtin(__builtin_amdgcn_tanhf)
    t = __builtin_amdgcn_tanhf(u);
#elif __has_builtin(__builtin_amdgcn_tanh_f32)
    t = __builtin_amdgcn_tanh_f32(u);
#else
    asm("v_tanh_f32 %0, %1" : "=v"(t) : "v"(u));   // CDNA5 TRANS op
#endif
    return 0.5f * x * (1.0f + t);
}

// ---------------------------------------------------------------------------
// Kernel 1: convert W1 ([131,128] f32, input-major) and W2 ([128,128] f32)
// to f16, TRANSPOSED to [N][K] (K padded to 160 for W1) so each WMMA
// B-fragment (two 8-half K runs per lane, 16 apart) is two aligned b128 LDS loads.
// ---------------------------------------------------------------------------
__global__ void prep_weights_f16(const float* __restrict__ W1,
                                 const float* __restrict__ W2,
                                 _Float16* __restrict__ W1t,   // [128][160]
                                 _Float16* __restrict__ W2t) { // [128][128]
    int idx = blockIdx.x * blockDim.x + threadIdx.x;
    const int total1 = D_FEAT * K1_PAD;   // 20480
    const int total2 = D_FEAT * D_FEAT;   // 16384
    if (idx < total1) {
        int n = idx / K1_PAD;
        int k = idx % K1_PAD;
        float v = (k < D_FEAT + DE_ATTR) ? W1[k * D_FEAT + n] : 0.0f;
        W1t[idx] = (_Float16)v;
    } else if (idx < total1 + total2) {
        int j = idx - total1;
        int n = j / D_FEAT;
        int k = j % D_FEAT;
        W2t[j] = (_Float16)W2[k * D_FEAT + n];
    }
}

// ---------------------------------------------------------------------------
// Kernel 2: acc = H   (so the scatter-add directly produces H + agg)
// ---------------------------------------------------------------------------
__global__ void init_acc(const float4* __restrict__ H4,
                         float4* __restrict__ acc4, int n4) {
    for (int i = blockIdx.x * blockDim.x + threadIdx.x; i < n4;
         i += gridDim.x * blockDim.x)
        acc4[i] = H4[i];
}

// ---------------------------------------------------------------------------
// Kernel 3: per-edge MLP on matrix cores + atomic scatter.
// 256 threads = 8 wave32s; block grid-strides over 32-edge tiles.
// Weights live in LDS for the block lifetime (72 KB). Each wave owns a
// 16-column slice and computes TWO 16x16 M-tiles per B-fragment load
// (18 WMMAs per loop body).
// ---------------------------------------------------------------------------
__launch_bounds__(256)
__global__ void edge_mlp_scatter(const float*     __restrict__ H,
                                 const long long* __restrict__ eidx,  // [2,E]
                                 const float*     __restrict__ eattr, // [E,3]
                                 const _Float16*  __restrict__ W1t,   // [128][160]
                                 const float*     __restrict__ b1,
                                 const _Float16*  __restrict__ W2t,   // [128][128]
                                 const float*     __restrict__ b2,
                                 float*           __restrict__ acc) {
    __shared__ __attribute__((aligned(16))) _Float16 sW1[D_FEAT * K1_PAD];  // 40 KB
    __shared__ __attribute__((aligned(16))) _Float16 sW2[D_FEAT * D_FEAT];  // 32 KB
    __shared__ __attribute__((aligned(16))) _Float16 sA [TILE_M * K1_PAD];  // 10 KB
    __shared__ __attribute__((aligned(16))) _Float16 sHd[TILE_M * D_FEAT];  //  8 KB
    __shared__ int   sDst[TILE_M];
    __shared__ float sB1[D_FEAT], sB2[D_FEAT];

    const int tid = threadIdx.x;

    // ---- one-time cooperative weight load into LDS (b128 chunks) ----
    {
        const uint4* g1 = (const uint4*)W1t; uint4* l1 = (uint4*)sW1;
        for (int i = tid; i < (D_FEAT * K1_PAD) / 8; i += 256) l1[i] = g1[i];
        const uint4* g2 = (const uint4*)W2t; uint4* l2 = (uint4*)sW2;
        for (int i = tid; i < (D_FEAT * D_FEAT) / 8; i += 256) l2[i] = g2[i];
        if (tid < D_FEAT) { sB1[tid] = b1[tid]; sB2[tid] = b2[tid]; }
    }
    __syncthreads();

    const long long* srcI = eidx;
    const long long* dstI = eidx + E_EDGES;

    const int lane = tid & 31;
    const int wid  = tid >> 5;        // wave 0..7
    const int lh   = lane & 15;       // matrix row / column within fragment
    const int khi  = lane >> 4;       // 0: lanes 0-15, 1: lanes 16-31
    const int aofs = khi * 8;         // A-frag K base (halves), per ISA layout
    const int bofs = khi * 8;         // B-frag K base (halves)
    const int ncol = wid * 16;        // this wave's output-column base

    const int numTiles = (E_EDGES + TILE_M - 1) / TILE_M;
    for (int tile = blockIdx.x; tile < numTiles; tile += gridDim.x) {
        const int base = tile * TILE_M;

        // ---- stage A tile: gather H[src] (f32->f16) + edge attrs + pad ----
        {
            const int s  = tid & 15;           // 8-float segment 0..15
            const int e0 = tid >> 4;           // edge slot 0..15
            #pragma unroll
            for (int eo = 0; eo < TILE_M; eo += 16) {
                const int e  = e0 + eo;
                const int eg = base + e;
                const bool valid = (eg < E_EDGES);
                const int  sidx  = valid ? (int)srcI[eg] : 0;

                const float4* hp = (const float4*)(H + (size_t)sidx * D_FEAT);
                float4 f0 = hp[s * 2 + 0];
                float4 f1 = hp[s * 2 + 1];
                union { _Float16 h[8]; uint4 u; } pk;
                pk.h[0] = (_Float16)f0.x; pk.h[1] = (_Float16)f0.y;
                pk.h[2] = (_Float16)f0.z; pk.h[3] = (_Float16)f0.w;
                pk.h[4] = (_Float16)f1.x; pk.h[5] = (_Float16)f1.y;
                pk.h[6] = (_Float16)f1.z; pk.h[7] = (_Float16)f1.w;
                if (!valid) pk.u = make_uint4(0u, 0u, 0u, 0u);
                *(uint4*)&sA[e * K1_PAD + s * 8] = pk.u;

                // K = 128..159: 3 edge attrs then zero pad (2 halves/thread)
                const int k0 = 128 + 2 * s;
                #pragma unroll
                for (int j = 0; j < 2; ++j) {
                    int   k = k0 + j;
                    float v = (valid && k < D_FEAT + DE_ATTR)
                                  ? eattr[(size_t)eg * DE_ATTR + (k - 128)]
                                  : 0.0f;
                    sA[e * K1_PAD + k] = (_Float16)v;
                }
            }
            if (tid < TILE_M) {
                int eg2 = base + tid;
                sDst[tid] = (eg2 < E_EDGES) ? (int)dstI[eg2] : -1;
            }
            if (tid == 0 && tile + gridDim.x < numTiles) {
                size_t nb = (size_t)(tile + gridDim.x) * TILE_M;
                __builtin_prefetch(&srcI[nb], 0, 0);   // global_prefetch_b8
                __builtin_prefetch(&dstI[nb], 0, 0);
            }
        }
        __syncthreads();

        // ---- GEMM1: [32 x 160] @ [160 x 16]: 10 WMMAs, B reused 2x ----
        v8f c1a = {}, c1b = {};
        #pragma unroll
        for (int kk = 0; kk < K1_PAD; kk += 32) {
            FragH b, a0, a1;
            const _Float16* bp = &sW1[(ncol + lh) * K1_PAD + kk + bofs];
            b.u[0]  = *(const uint4*)(bp);
            b.u[1]  = *(const uint4*)(bp + 16);
            const _Float16* ap0 = &sA[lh * K1_PAD + kk + aofs];
            a0.u[0] = *(const uint4*)(ap0);
            a0.u[1] = *(const uint4*)(ap0 + 16);
            const _Float16* ap1 = &sA[(lh + 16) * K1_PAD + kk + aofs];
            a1.u[0] = *(const uint4*)(ap1);
            a1.u[1] = *(const uint4*)(ap1 + 16);
            c1a = __builtin_amdgcn_wmma_f32_16x16x32_f16(
                      false, a0.h, false, b.h, (short)0, c1a, false, false);
            c1b = __builtin_amdgcn_wmma_f32_16x16x32_f16(
                      false, a1.h, false, b.h, (short)0, c1b, false, false);
        }
        // bias + GELU (hardware tanh) -> f16 activation tile in LDS
        {
            const int   col = ncol + lh;
            const float bb  = sB1[col];
            #pragma unroll
            for (int r = 0; r < 8; ++r) {
                int m = r + khi * 8;
                sHd[m * D_FEAT + col]        = (_Float16)gelu_fast(c1a[r] + bb);
                sHd[(m + 16) * D_FEAT + col] = (_Float16)gelu_fast(c1b[r] + bb);
            }
        }
        __syncthreads();

        // ---- GEMM2: [32 x 128] @ [128 x 16]: 8 WMMAs, B reused 2x ----
        v8f c2a = {}, c2b = {};
        #pragma unroll
        for (int kk = 0; kk < D_FEAT; kk += 32) {
            FragH b, a0, a1;
            const _Float16* bp = &sW2[(ncol + lh) * D_FEAT + kk + bofs];
            b.u[0]  = *(const uint4*)(bp);
            b.u[1]  = *(const uint4*)(bp + 16);
            const _Float16* ap0 = &sHd[lh * D_FEAT + kk + aofs];
            a0.u[0] = *(const uint4*)(ap0);
            a0.u[1] = *(const uint4*)(ap0 + 16);
            const _Float16* ap1 = &sHd[(lh + 16) * D_FEAT + kk + aofs];
            a1.u[0] = *(const uint4*)(ap1);
            a1.u[1] = *(const uint4*)(ap1 + 16);
            c2a = __builtin_amdgcn_wmma_f32_16x16x32_f16(
                      false, a0.h, false, b.h, (short)0, c2a, false, false);
            c2b = __builtin_amdgcn_wmma_f32_16x16x32_f16(
                      false, a1.h, false, b.h, (short)0, c2b, false, false);
        }
        // bias + atomic scatter-add into the (L2-resident) accumulator
        {
            const int   col = ncol + lh;
            const float bb  = sB2[col];
            #pragma unroll
            for (int r = 0; r < 8; ++r) {
                int m  = r + khi * 8;
                int d0 = sDst[m];
                int d1 = sDst[m + 16];
                if (d0 >= 0)
                    atomicAdd(&acc[(size_t)d0 * D_FEAT + col], c2a[r] + bb);
                if (d1 >= 0)
                    atomicAdd(&acc[(size_t)d1 * D_FEAT + col], c2b[r] + bb);
            }
        }
        __syncthreads();   // LDS tiles reused next iteration
    }
}

// ---------------------------------------------------------------------------
// Kernel 4: row LayerNorm; one wave32 per node row, float4 lanes, shfl_xor.
// ---------------------------------------------------------------------------
__global__ void layernorm_out(const float* __restrict__ acc,
                              const float* __restrict__ gamma,
                              const float* __restrict__ beta,
                              float* __restrict__ out) {
    const int row  = blockIdx.x * 8 + (threadIdx.x >> 5);
    const int lane = threadIdx.x & 31;
    if (row >= T_NODES) return;

    const float4* rp = (const float4*)(acc + (size_t)row * D_FEAT);
    float4 v = rp[lane];
    float  s = v.x + v.y + v.z + v.w;
    float  q = v.x * v.x + v.y * v.y + v.z * v.z + v.w * v.w;
    #pragma unroll
    for (int m = 16; m >= 1; m >>= 1) {
        s += __shfl_xor(s, m, 32);
        q += __shfl_xor(q, m, 32);
    }
    const float mu  = s * (1.0f / D_FEAT);
    const float var = q * (1.0f / D_FEAT) - mu * mu;
    const float inv = rsqrtf(var + LN_EPS);

    float4 g = ((const float4*)gamma)[lane];
    float4 b = ((const float4*)beta )[lane];
    float4 o;
    o.x = (v.x - mu) * inv * g.x + b.x;
    o.y = (v.y - mu) * inv * g.y + b.y;
    o.z = (v.z - mu) * inv * g.z + b.z;
    o.w = (v.w - mu) * inv * g.w + b.w;
    ((float4*)(out + (size_t)row * D_FEAT))[lane] = o;
}

// ---------------------------------------------------------------------------
extern "C" void kernel_launch(void* const* d_in, const int* in_sizes, int n_in,
                              void* d_out, int out_size, void* d_ws, size_t ws_size,
                              hipStream_t stream) {
    const float*     H     = (const float*)d_in[0];
    const long long* eidx  = (const long long*)d_in[1];   // int64 [2,E]
    const float*     eattr = (const float*)d_in[2];
    const float*     W1    = (const float*)d_in[3];
    const float*     b1    = (const float*)d_in[4];
    const float*     W2    = (const float*)d_in[5];
    const float*     b2    = (const float*)d_in[6];
    const float*     gamma = (const float*)d_in[7];
    const float*     beta  = (const float*)d_in[8];
    float*           out   = (float*)d_out;

    // workspace layout: f32 accumulator [T,D] then f16 transposed weights
    char*     ws  = (char*)d_ws;
    float*    acc = (float*)ws;
    _Float16* W1t = (_Float16*)(ws + (size_t)T_NODES * D_FEAT * sizeof(float));
    _Float16* W2t = W1t + D_FEAT * K1_PAD;

    const int wtotal = D_FEAT * K1_PAD + D_FEAT * D_FEAT;
    prep_weights_f16<<<(wtotal + 255) / 256, 256, 0, stream>>>(W1, W2, W1t, W2t);
    init_acc<<<2048, 256, 0, stream>>>((const float4*)H, (float4*)acc,
                                       T_NODES * D_FEAT / 4);
    edge_mlp_scatter<<<2048, 256, 0, stream>>>(H, eidx, eattr, W1t, b1, W2t, b2, acc);
    layernorm_out<<<(T_NODES + 7) / 8, 256, 0, stream>>>(acc, gamma, beta, out);
}
